// GraphOfShotsV2_53017076302394
// MI455X (gfx1250) — compile-verified, matrix operand order checked by reading, not run on hardware
//
#include <hip/hip_runtime.h>

#define NS_   4096
#define NQ_   4096
#define NTOT  8192
#define DD    256
#define NWAY_ 64
#define METAK 32
#define MIN_  320   // D + NWAY
#define REF_  2

typedef __bf16 bf16;
typedef __attribute__((ext_vector_type(16))) __bf16 v16bf;
typedef __attribute__((ext_vector_type(8)))  float  v8f;
typedef __attribute__((ext_vector_type(4)))  unsigned int v4u;
typedef __attribute__((ext_vector_type(8)))  int  v8i;
typedef __attribute__((ext_vector_type(4)))  int  v4i;

// ---------------- WMMA fragment helpers (wave32, 16x16x32 bf16) ----------------
// A (16xK tile, row-major source): lane m = lane&15, half = lane>>4.
// elements 0..7 -> K = half*8 + e ; elements 8..15 -> K = 16 + half*8 + (e-8)
__device__ inline v16bf load_fragA(const bf16* __restrict__ base, int ld) {
  int lane = threadIdx.x & 31;
  int m = lane & 15, half = lane >> 4;
  const bf16* rp = base + (size_t)m * ld + half * 8;
  v16bf f;
#pragma unroll
  for (int e = 0; e < 8; ++e) f[e] = rp[e];
#pragma unroll
  for (int e = 0; e < 8; ++e) f[8 + e] = rp[16 + e];
  return f;
}

// B (KxN tile, K-major source: B[k][n] = base[k*ld + n]); lane n = lane&15
__device__ inline v16bf load_fragB_km(const bf16* __restrict__ base, int ld) {
  int lane = threadIdx.x & 31;
  int n = lane & 15, half = lane >> 4;
  v16bf f;
#pragma unroll
  for (int e = 0; e < 8; ++e) f[e] = base[(size_t)(half * 8 + e) * ld + n];
#pragma unroll
  for (int e = 0; e < 8; ++e) f[8 + e] = base[(size_t)(16 + half * 8 + e) * ld + n];
  return f;
}

// ---------------- TDM: stage a 2D bf16 tile (tile_rows x cols, row-major,
// row stride == cols) from global into LDS. D# per CDNA5 ISA.
// This toolchain's builtin takes 6 args (g0, g1, g2, g3, extra group, cpol). ----------------
__device__ inline void tdm_load_2d_bf16(unsigned int lds_off, const void* gptr,
                                        unsigned int rows_total, unsigned int cols,
                                        unsigned int tile_rows) {
  unsigned long long ga = (unsigned long long)(size_t)gptr;
  v4u g0;
  g0.x = 1u;                                     // count=1, is_restore=0, gather=0
  g0.y = lds_off;                                // lds_addr
  g0.z = (unsigned int)ga;                       // global_addr[31:0]
  g0.w = (unsigned int)((ga >> 32) & 0x01FFFFFFull) | (2u << 30);  // addr[56:32], type=2
  v8i g1;
  g1[0] = (int)(1u << 16);                       // workgroup_mask=0, data_size=1 (2B)
  g1[1] = (int)((cols & 0xFFFFu) << 16);         // tensor_dim0[15:0]
  g1[2] = (int)((cols >> 16) | ((rows_total & 0xFFFFu) << 16));   // dim0 hi | dim1 lo
  g1[3] = (int)((rows_total >> 16) | ((cols & 0xFFFFu) << 16));   // dim1 hi | tile_dim0
  g1[4] = (int)(tile_rows & 0xFFFFu);            // tile_dim1 ; tile_dim2=0
  g1[5] = (int)cols;                             // tensor_dim0_stride[31:0]
  g1[6] = 0;                                     // stride hi, dim1_stride lo
  g1[7] = 0;
  v4i z4 = {0, 0, 0, 0};                         // groups 2/3 unused (2D tensor)
  v8i z8 = {0, 0, 0, 0, 0, 0, 0, 0};             // extra group (clang-23 6-arg form)
  __builtin_amdgcn_tensor_load_to_lds(g0, g1, z4, z4, z8, 0);
}

// ---------------- elementwise / reduction kernels ----------------
__global__ void f2bf_kernel(const float* __restrict__ in, bf16* __restrict__ out, int n) {
  int i = blockIdx.x * blockDim.x + threadIdx.x;
  if (i < n) out[i] = (bf16)in[i];
}

// row L2-normalize (D=256) -> bf16
__global__ void rownorm_bf16_kernel(const float* __restrict__ x, bf16* __restrict__ out) {
  __shared__ float sh[DD];
  int row = blockIdx.x, t = threadIdx.x;
  float v = x[(size_t)row * DD + t];
  sh[t] = v * v;
  __syncthreads();
  for (int o = 128; o > 0; o >>= 1) { if (t < o) sh[t] += sh[t + o]; __syncthreads(); }
  float rs = rsqrtf(sh[0] + 1e-12f);
  out[(size_t)row * DD + t] = (bf16)(v * rs);
}

// build GCN layer-1 input: [z | one_hot(label)] as bf16, width 320
__global__ void build_xb_kernel(const float* __restrict__ z, const int* __restrict__ slab,
                                bf16* __restrict__ xb) {
  int row = blockIdx.x, c = threadIdx.x;
  float v;
  if (c < DD) v = z[(size_t)row * DD + c];
  else {
    v = 0.f;
    if (row < NS_ && slab[row] == (c - DD)) v = 1.f;
  }
  xb[(size_t)row * MIN_ + c] = (bf16)v;
}

// ---------------- fused affinity GEMM + top-32 ----------------
// one wave per block; block owns 32 rows (two 16-row WMMA strips).
// lane L owns row rowBase+L's top-32 list in LDS.
__global__ __launch_bounds__(32) void aff_topk_kernel(const bf16* __restrict__ zb,
                                                      float* __restrict__ topv,
                                                      int* __restrict__ topi) {
  __shared__ float stage[16][17];
  __shared__ float lv[32][METAK];
  __shared__ int   li[32][METAK];
  int lane = threadIdx.x;
  int rowBase = blockIdx.x * 32;
#pragma unroll
  for (int k = 0; k < METAK; ++k) { lv[lane][k] = -3.0e38f; li[lane][k] = 0; }
  float curmin = -3.0e38f; int minpos = 0;
  __syncthreads();
  for (int s = 0; s < 2; ++s) {
    int m0 = rowBase + s * 16;
    v16bf afrag[8];
#pragma unroll
    for (int kk = 0; kk < 8; ++kk) afrag[kk] = load_fragA(zb + (size_t)m0 * DD + kk * 32, DD);
    bool owner = ((lane >> 4) == s);
    int mloc = lane & 15;
    int half = lane >> 4, cc = lane & 15;
    for (int j0 = 0; j0 < NTOT; j0 += 16) {
      if (j0 + 16 < NTOT)  // hide next panel's latency behind this tile's WMMAs
        __builtin_prefetch(zb + (size_t)(j0 + 16 + lane) * DD, 0, 1);
      // preload all B fragments so loads overlap the WMMA chain
      v16bf bfr[8];
#pragma unroll
      for (int kk = 0; kk < 8; ++kk)
        bfr[kk] = load_fragA(zb + (size_t)j0 * DD + kk * 32, DD);  // symmetric B
      v8f acc = {};
#pragma unroll
      for (int kk = 0; kk < 8; ++kk)
        acc = __builtin_amdgcn_wmma_f32_16x16x32_bf16(false, afrag[kk], false, bfr[kk],
                                                      (short)0, acc, false, false);
#pragma unroll
      for (int r = 0; r < 8; ++r) stage[r + 8 * half][cc] = acc[r];
      __syncthreads();
      if (owner) {
        int i = m0 + mloc;
        for (int c = 0; c < 16; ++c) {
          int j = j0 + c;
          float v = stage[mloc][c];
          if (j == i) v -= 2.0f;
          if (v > curmin) {
            lv[lane][minpos] = v; li[lane][minpos] = j;
            float mn = lv[lane][0]; int mp = 0;
#pragma unroll
            for (int k = 1; k < METAK; ++k) { float t = lv[lane][k]; if (t < mn) { mn = t; mp = k; } }
            curmin = mn; minpos = mp;
          }
        }
      }
      __syncthreads();
    }
  }
  __syncthreads();
  // sort own row descending (tie -> lower index), write out
  int i = rowBase + lane;
  for (int a = 0; a < METAK; ++a) {
    int best = a;
    for (int b = a + 1; b < METAK; ++b) {
      if (lv[lane][b] > lv[lane][best] ||
          (lv[lane][b] == lv[lane][best] && li[lane][b] < li[lane][best])) best = b;
    }
    float tv = lv[lane][a]; lv[lane][a] = lv[lane][best]; lv[lane][best] = tv;
    int ti = li[lane][a]; li[lane][a] = li[lane][best]; li[lane][best] = ti;
    topv[(size_t)i * METAK + a] = lv[lane][a];
    topi[(size_t)i * METAK + a] = li[lane][a];
  }
}

// ---------------- WMMA GEMM: out(N x 256) = Xb(N x KIN) @ Wb(KIN x 256) ----------------
// A panel (16 x KIN) staged into LDS via TDM tensor_load_to_lds.
template <int KIN>
__global__ __launch_bounds__(32) void gemm_xw_kernel(const bf16* __restrict__ X,
                                                     const bf16* __restrict__ W,
                                                     float* __restrict__ out) {
  __shared__ bf16 shA[16 * KIN];
  int m0 = blockIdx.x * 16;
  int n0 = blockIdx.y * 64;
  // async DMA of the A panel into LDS (one TDM descriptor per wave)
  tdm_load_2d_bf16((unsigned int)(size_t)(void*)shA, X + (size_t)m0 * KIN,
                   (unsigned int)NTOT, (unsigned int)KIN, 16u);
  __builtin_amdgcn_s_wait_tensorcnt(0);
  __syncthreads();
  v8f acc[4] = {};
  for (int k0 = 0; k0 < KIN; k0 += 32) {
    v16bf a = load_fragA(shA + k0, KIN);
    v16bf b[4];
#pragma unroll
    for (int t = 0; t < 4; ++t)
      b[t] = load_fragB_km(W + (size_t)k0 * DD + n0 + t * 16, DD);
#pragma unroll
    for (int t = 0; t < 4; ++t)
      acc[t] = __builtin_amdgcn_wmma_f32_16x16x32_bf16(false, a, false, b[t],
                                                       (short)0, acc[t], false, false);
  }
  int lane = threadIdx.x, cc = lane & 15, half = lane >> 4;
#pragma unroll
  for (int t = 0; t < 4; ++t)
#pragma unroll
    for (int r = 0; r < 8; ++r)
      out[(size_t)(m0 + r + 8 * half) * DD + n0 + t * 16 + cc] = acc[t][r];
}

// degree / dinv : deg[i] = 1 (self loop) + sum_k w_ik ; w zeroed for src >= NS
__global__ void dinv_kernel(const float* __restrict__ topv, const int* __restrict__ topi,
                            float* __restrict__ dinv) {
  int i = blockIdx.x * blockDim.x + threadIdx.x;
  if (i >= NTOT) return;
  float deg = 1.0f;
#pragma unroll
  for (int k = 0; k < METAK; ++k) {
    float w = (topv[i * METAK + k] + 1.f) * 0.5f;
    if (topi[i * METAK + k] >= NS_) w = 0.f;
    deg += w;
  }
  dinv[i] = rsqrtf(fmaxf(deg, 1e-12f));
}

// GCN aggregation (row-local gather; dst = own top-32 + self loop)
__global__ void gcn_agg_kernel(const float* __restrict__ xw, const float* __restrict__ topv,
                               const int* __restrict__ topi, const float* __restrict__ dinv,
                               const float* __restrict__ bias, float* __restrict__ out) {
  int i = blockIdx.x, d = threadIdx.x;
  float di = dinv[i];
  float acc = di * di * xw[(size_t)i * DD + d];
  for (int k = 0; k < METAK; ++k) {
    int s = topi[i * METAK + k];
    float w = (topv[i * METAK + k] + 1.f) * 0.5f;
    if (s >= NS_) w = 0.f;
    acc += dinv[s] * w * di * xw[(size_t)s * DD + d];
  }
  out[(size_t)i * DD + d] = acc + bias[d];
}

// batch-norm column stats over NTOT rows (biased var)
__global__ void colstats_kernel(const float* __restrict__ x, float* __restrict__ mv) {
  __shared__ float sh[256], sh2[256];
  int d = blockIdx.x, t = threadIdx.x;
  float s = 0.f, s2 = 0.f;
  for (int r = t; r < NTOT; r += 256) { float v = x[(size_t)r * DD + d]; s += v; s2 += v * v; }
  sh[t] = s; sh2[t] = s2;
  __syncthreads();
  for (int o = 128; o > 0; o >>= 1) {
    if (t < o) { sh[t] += sh[t + o]; sh2[t] += sh2[t + o]; }
    __syncthreads();
  }
  if (t == 0) {
    float m = sh[0] / NTOT;
    mv[d] = m;
    mv[DD + d] = sh2[0] / NTOT - m * m;
  }
}

__global__ void bn_kernel(const float* __restrict__ x, const float* __restrict__ mv,
                          const float* __restrict__ g, const float* __restrict__ b,
                          float* __restrict__ out, int dorelu) {
  int i = blockIdx.x, d = threadIdx.x;
  float y = (x[(size_t)i * DD + d] - mv[d]) * rsqrtf(mv[DD + d] + 1e-5f) * g[d] + b[d];
  if (dorelu) y = fmaxf(y, 0.f);
  out[(size_t)i * DD + d] = y;
}

// alpha-blended class prototypes (deterministic scan, no atomics)
__global__ void protos_kernel(const float* __restrict__ h_s, const float* __restrict__ z,
                              const int* __restrict__ slab, const float* __restrict__ alog,
                              float* __restrict__ protos) {
  int lab = blockIdx.x, d = threadIdx.x;
  float ah = 0.f, az = 0.f; int cnt = 0;
  for (int r = 0; r < NS_; ++r) {
    if (slab[r] == lab) { ah += h_s[(size_t)r * DD + d]; az += z[(size_t)r * DD + d]; ++cnt; }
  }
  float c = fmaxf((float)cnt, 1.f);
  float alpha = 1.f / (1.f + __expf(-alog[0]));
  protos[(size_t)lab * DD + d] = alpha * (ah / c) + (1.f - alpha) * (az / c);
}

__global__ void mix_kernel(const float* __restrict__ h_s, const float* __restrict__ h_q,
                           const float* __restrict__ z, const float* __restrict__ alog,
                           float* __restrict__ zsmix, float* __restrict__ zqmix) {
  int i = blockIdx.x, d = threadIdx.x;
  float alpha = 1.f / (1.f + __expf(-alog[0]));
  if (i < NS_) {
    zsmix[(size_t)i * DD + d] = alpha * h_s[(size_t)i * DD + d] + (1.f - alpha) * z[(size_t)i * DD + d];
  } else {
    int q = i - NS_;
    zqmix[(size_t)q * DD + d] = alpha * h_q[(size_t)q * DD + d] + (1.f - alpha) * z[(size_t)i * DD + d];
  }
}

// logits (sqrt-euclidean to prototypes) + per-row CE
__global__ void logits_ce_kernel(const float* __restrict__ zqmix, const float* __restrict__ protos,
                                 const int* __restrict__ qlab, float* __restrict__ logits,
                                 float* __restrict__ ce_row) {
  __shared__ float sh[NWAY_];
  __shared__ float labL;
  int q = blockIdx.x, t = threadIdx.x;
  const float* zr = zqmix + (size_t)q * DD;
  const float* pr = protos + (size_t)t * DD;
  float qq = 0.f, pp = 0.f, qp = 0.f;
  for (int k = 0; k < DD; ++k) { float a = zr[k], b = pr[k]; qq += a * a; pp += b * b; qp += a * b; }
  float lg = -sqrtf(fmaxf(qq + pp - 2.f * qp, 1e-12f));
  logits[(size_t)q * NWAY_ + t] = lg;
  if (t == qlab[q]) labL = lg;
  sh[t] = lg;
  __syncthreads();
  for (int o = 32; o > 0; o >>= 1) { if (t < o) sh[t] = fmaxf(sh[t], sh[t + o]); __syncthreads(); }
  float mx = sh[0];
  __syncthreads();
  sh[t] = __expf(lg - mx);
  __syncthreads();
  for (int o = 32; o > 0; o >>= 1) { if (t < o) sh[t] += sh[t + o]; __syncthreads(); }
  if (t == 0) ce_row[q] = (mx + __logf(sh[0])) - labL;
}

// fused sim GEMM (WMMA) + streaming masked/unmasked logsumexp per query row
__global__ __launch_bounds__(32) void sim_con_kernel(const bf16* __restrict__ qb,
                                                     const bf16* __restrict__ sb,
                                                     const int* __restrict__ qlab,
                                                     const int* __restrict__ slab,
                                                     float* __restrict__ con_row,
                                                     float* __restrict__ valid_row) {
  __shared__ float stage[16][17];
  __shared__ int slabsh[16];
  int lane = threadIdx.x;
  int qBase = blockIdx.x * 32;
  float mA = -3e38f, sA = 0.f, mN = -3e38f, sN = 0.f;
  int myq = qBase + lane;
  int myl = qlab[myq];
  const float invT = 10.0f;  // 1/TEMP
  for (int s = 0; s < 2; ++s) {
    int m0 = qBase + s * 16;
    v16bf afrag[8];
#pragma unroll
    for (int kk = 0; kk < 8; ++kk) afrag[kk] = load_fragA(qb + (size_t)m0 * DD + kk * 32, DD);
    bool owner = ((lane >> 4) == s);
    int mloc = lane & 15, half = lane >> 4, cc = lane & 15;
    for (int j0 = 0; j0 < NS_; j0 += 16) {
      if (j0 + 16 < NS_)
        __builtin_prefetch(sb + (size_t)(j0 + 16 + lane) * DD, 0, 1);
      v16bf bfr[8];
#pragma unroll
      for (int kk = 0; kk < 8; ++kk)
        bfr[kk] = load_fragA(sb + (size_t)j0 * DD + kk * 32, DD);
      v8f acc = {};
#pragma unroll
      for (int kk = 0; kk < 8; ++kk)
        acc = __builtin_amdgcn_wmma_f32_16x16x32_bf16(false, afrag[kk], false, bfr[kk],
                                                      (short)0, acc, false, false);
      if (lane < 16) slabsh[lane] = slab[j0 + lane];
#pragma unroll
      for (int r = 0; r < 8; ++r) stage[r + 8 * half][cc] = acc[r];
      __syncthreads();
      if (owner) {
        for (int c = 0; c < 16; ++c) {
          float v = stage[mloc][c] * invT;
          if (v > mA) { sA = sA * __expf(mA - v) + 1.f; mA = v; }
          else sA += __expf(v - mA);
          if (slabsh[c] == myl) {
            if (v > mN) { sN = sN * __expf(mN - v) + 1.f; mN = v; }
            else sN += __expf(v - mN);
          }
        }
      }
      __syncthreads();
    }
  }
  if (sN > 0.f) { con_row[myq] = (mN + __logf(sN)) - (mA + __logf(sA)); valid_row[myq] = 1.f; }
  else { con_row[myq] = 0.f; valid_row[myq] = 0.f; }
}

// deterministic final reduction: loss = mean(ce) + 0.5 * (-sum(con)/max(nv,1))
__global__ void final_loss_kernel(const float* __restrict__ ce_row, const float* __restrict__ con_row,
                                  const float* __restrict__ valid_row, float* __restrict__ out_loss) {
  __shared__ float shc[256], shn[256], shv[256];
  int t = threadIdx.x;
  float sc = 0.f, sn = 0.f, sv = 0.f;
  for (int i = t; i < NQ_; i += 256) { sc += ce_row[i]; sn += con_row[i]; sv += valid_row[i]; }
  shc[t] = sc; shn[t] = sn; shv[t] = sv;
  __syncthreads();
  for (int o = 128; o > 0; o >>= 1) {
    if (t < o) { shc[t] += shc[t + o]; shn[t] += shn[t + o]; shv[t] += shv[t + o]; }
    __syncthreads();
  }
  if (t == 0) {
    float ce = shc[0] / (float)NQ_;
    float con = -shn[0] / fmaxf(shv[0], 1.f);
    out_loss[0] = ce + 0.5f * con;
  }
}

// ---------------- host orchestration ----------------
extern "C" void kernel_launch(void* const* d_in, const int* in_sizes, int n_in,
                              void* d_out, int out_size, void* d_ws, size_t ws_size,
                              hipStream_t stream) {
  (void)in_sizes; (void)n_in; (void)out_size; (void)ws_size;
  const float* h_s  = (const float*)d_in[0];
  const float* h_q  = (const float*)d_in[1];
  const int*   slab = (const int*)d_in[2];
  const int*   qlab = (const int*)d_in[3];
  const float* W1   = (const float*)d_in[4];
  const float* b1   = (const float*)d_in[5];
  const float* g1   = (const float*)d_in[6];
  const float* be1  = (const float*)d_in[7];
  const float* W2   = (const float*)d_in[8];
  const float* b2   = (const float*)d_in[9];
  const float* g2   = (const float*)d_in[10];
  const float* be2  = (const float*)d_in[11];
  const float* alog = (const float*)d_in[12];
  float* out = (float*)d_out;   // [NQ*NWAY logits][1 loss]

  char* p = (char*)d_ws;
  auto alloc = [&](size_t bytes) -> void* {
    void* r = (void*)p;
    p += (bytes + 255) & ~(size_t)255;
    return r;
  };
  float* z     = (float*)alloc((size_t)NTOT * DD * 4);
  bf16*  zb    = (bf16*)alloc((size_t)NTOT * DD * 2);
  bf16*  xb    = (bf16*)alloc((size_t)NTOT * MIN_ * 2);   // also reused as bf16(h)
  float* xw    = (float*)alloc((size_t)NTOT * DD * 4);
  float* h     = (float*)alloc((size_t)NTOT * DD * 4);
  float* topv  = (float*)alloc((size_t)NTOT * METAK * 4);
  int*   topi  = (int*)alloc((size_t)NTOT * METAK * 4);
  float* dinv  = (float*)alloc((size_t)NTOT * 4);
  float* mv    = (float*)alloc((size_t)2 * DD * 4);
  bf16*  wb1   = (bf16*)alloc((size_t)REF_ * MIN_ * DD * 2);
  bf16*  wb2   = (bf16*)alloc((size_t)REF_ * DD * DD * 2);
  float* protos= (float*)alloc((size_t)NWAY_ * DD * 4);
  float* zsmix = (float*)alloc((size_t)NS_ * DD * 4);
  float* zqmix = (float*)alloc((size_t)NQ_ * DD * 4);
  bf16*  sb    = (bf16*)alloc((size_t)NS_ * DD * 2);
  bf16*  qb    = (bf16*)alloc((size_t)NQ_ * DD * 2);
  float* ce_row   = (float*)alloc((size_t)NQ_ * 4);
  float* con_row  = (float*)alloc((size_t)NQ_ * 4);
  float* valid_row= (float*)alloc((size_t)NQ_ * 4);

  // z = concat(h_s, h_q)
  (void)hipMemcpyAsync(z, h_s, (size_t)NS_ * DD * 4, hipMemcpyDeviceToDevice, stream);
  (void)hipMemcpyAsync(z + (size_t)NS_ * DD, h_q, (size_t)NQ_ * DD * 4, hipMemcpyDeviceToDevice, stream);

  // weights -> bf16 once
  {
    int n1 = REF_ * MIN_ * DD, n2 = REF_ * DD * DD;
    f2bf_kernel<<<(n1 + 255) / 256, 256, 0, stream>>>(W1, wb1, n1);
    f2bf_kernel<<<(n2 + 255) / 256, 256, 0, stream>>>(W2, wb2, n2);
  }

  for (int r = 0; r < REF_; ++r) {
    // zn -> bf16, fused affinity + top-32
    rownorm_bf16_kernel<<<NTOT, DD, 0, stream>>>(z, zb);
    aff_topk_kernel<<<NTOT / 32, 32, 0, stream>>>(zb, topv, topi);
    dinv_kernel<<<NTOT / 256, 256, 0, stream>>>(topv, topi, dinv);

    // GCN layer 1: [z | onehot] @ W1[r], aggregate, BN+ReLU
    build_xb_kernel<<<NTOT, MIN_, 0, stream>>>(z, slab, xb);
    gemm_xw_kernel<MIN_><<<dim3(NTOT / 16, 4), 32, 0, stream>>>(xb, wb1 + (size_t)r * MIN_ * DD, xw);
    gcn_agg_kernel<<<NTOT, DD, 0, stream>>>(xw, topv, topi, dinv, b1 + r * DD, h);
    colstats_kernel<<<DD, 256, 0, stream>>>(h, mv);
    bn_kernel<<<NTOT, DD, 0, stream>>>(h, mv, g1 + r * DD, be1 + r * DD, h, 1);

    // GCN layer 2: h @ W2[r], aggregate, BN (no relu) -> z
    f2bf_kernel<<<(NTOT * DD + 255) / 256, 256, 0, stream>>>(h, xb, NTOT * DD);
    gemm_xw_kernel<DD><<<dim3(NTOT / 16, 4), 32, 0, stream>>>(xb, wb2 + (size_t)r * DD * DD, xw);
    gcn_agg_kernel<<<NTOT, DD, 0, stream>>>(xw, topv, topi, dinv, b2 + r * DD, z);
    colstats_kernel<<<DD, 256, 0, stream>>>(z, mv);
    bn_kernel<<<NTOT, DD, 0, stream>>>(z, mv, g2 + r * DD, be2 + r * DD, z, 0);
  }

  // prototypes and mixes
  protos_kernel<<<NWAY_, DD, 0, stream>>>(h_s, z, slab, alog, protos);
  mix_kernel<<<NTOT, DD, 0, stream>>>(h_s, h_q, z, alog, zsmix, zqmix);

  // logits + CE rows (logits straight to d_out)
  logits_ce_kernel<<<NQ_, NWAY_, 0, stream>>>(zqmix, protos, qlab, out, ce_row);

  // contrastive: normalize mixes -> bf16, fused sim + streaming logsumexp
  rownorm_bf16_kernel<<<NQ_, DD, 0, stream>>>(zqmix, qb);
  rownorm_bf16_kernel<<<NS_, DD, 0, stream>>>(zsmix, sb);
  sim_con_kernel<<<NQ_ / 32, 32, 0, stream>>>(qb, sb, qlab, slab, con_row, valid_row);

  // final scalar loss
  final_loss_kernel<<<1, 256, 0, stream>>>(ce_row, con_row, valid_row, out + (size_t)NQ_ * NWAY_);
}
